// CrossTransformerBlock_NoFine_31310311588317
// MI455X (gfx1250) — compile-verified, matrix operand order checked by reading, not run on hardware
//
#include <hip/hip_runtime.h>

// ---------------- problem constants (fixed by the reference) ----------------
#define DIM   256
#define NPTS  2048
#define NQ    4096
#define BATCH 4
#define KNN   16
#define QT    4          // queries per fused block

typedef __bf16 bf16;
typedef __attribute__((ext_vector_type(16))) bf16  v16bf;
typedef __attribute__((ext_vector_type(8)))  bf16  v8bf;
typedef __attribute__((ext_vector_type(8)))  float v8f;

// ---------------- helpers ----------------
__device__ __forceinline__ bf16 tobf(float f) {
  unsigned u = __builtin_bit_cast(unsigned, f);
  u += 0x7FFFu + ((u >> 16) & 1u);           // round-to-nearest-even
  unsigned short s = (unsigned short)(u >> 16);
  return __builtin_bit_cast(bf16, s);
}
__device__ __forceinline__ float bf2f(bf16 v) {
  unsigned short s = __builtin_bit_cast(unsigned short, v);
  unsigned u = ((unsigned)s) << 16;
  return __builtin_bit_cast(float, u);
}

// gfx1250 async global->LDS copy (16B per lane), tracked by ASYNCcnt.
// Generic LDS pointer: low 32 bits are the LDS byte address (aperture layout).
__device__ __forceinline__ void async_copy_b128(void* lds_ptr, const void* gptr) {
  unsigned loff = (unsigned)(size_t)lds_ptr;
  asm volatile("global_load_async_to_lds_b128 %0, %1, off"
               :: "v"(loff), "v"(gptr) : "memory");
}
__device__ __forceinline__ void wait_async0() {
  asm volatile("s_wait_asynccnt 0" ::: "memory");
}

// A fragment (16x32, M x K) from a row-major [rows][256] bf16 buffer, and
// B fragment (32x16, K x N) for X @ W^T from row-major W[n][k]: identical
// per-lane addressing (ISA 7.12.2): elements e<8 -> k = kk + 8*half + e,
// e>=8 -> k = kk + 16 + 8*half + (e-8); row/col = lane&15.
__device__ __forceinline__ v16bf load_frag(const bf16* __restrict__ base, int row0, int kk) {
  const int lane = threadIdx.x & 31;
  const int half = lane >> 4;
  const bf16* p = base + (row0 + (lane & 15)) * DIM + kk + half * 8;
  v8bf lo = *(const v8bf*)(p);
  v8bf hi = *(const v8bf*)(p + 16);
  v16bf a;
#pragma unroll
  for (int i = 0; i < 8; ++i) { a[i] = lo[i]; a[i + 8] = hi[i]; }
  return a;
}

__device__ __forceinline__ v8f wmma_acc(v16bf a, v16bf b, v8f c) {
  return __builtin_amdgcn_wmma_f32_16x16x32_bf16(false, a, false, b, (short)0, c, false, false);
}

// ---------------- kernel 1: f32 -> bf16 conversion (weights + points) ----------------
__global__ void conv_bf16_kernel(const float* __restrict__ Wd2, const float* __restrict__ Wg1,
                                 const float* __restrict__ Wg2, const float* __restrict__ Wk,
                                 const float* __restrict__ Wv, const float* __restrict__ pts,
                                 bf16* __restrict__ wdst, bf16* __restrict__ pdst) {
  const int per = DIM * DIM;
  const int wtot = 5 * per;
  const int ptot = BATCH * NPTS * DIM;
  for (int e = blockIdx.x * blockDim.x + threadIdx.x; e < wtot + ptot; e += gridDim.x * blockDim.x) {
    if (e < wtot) {
      int m = e / per, o = e % per;
      const float* src = (m == 0) ? Wd2 : (m == 1) ? Wg1 : (m == 2) ? Wg2 : (m == 3) ? Wk : Wv;
      wdst[e] = tobf(src[o]);
    } else {
      int o = e - wtot;
      pdst[o] = tobf(pts[o]);
    }
  }
}

// ---------------- kernel 2: q_attn / k_global / v_global (tiny GEMVs) ----------------
__global__ void small_proj_kernel(const float* __restrict__ lat, const float* __restrict__ Wq,
                                  const float* __restrict__ Wkg, const float* __restrict__ Wvg,
                                  float* __restrict__ qa, float* __restrict__ kg, float* __restrict__ vg) {
  int mat = blockIdx.x % 3, bb = blockIdx.x / 3;
  const float* W = (mat == 0) ? Wq : (mat == 1) ? Wkg : Wvg;
  float* out = (mat == 0) ? qa : (mat == 1) ? kg : vg;
  const float* l = lat + bb * DIM;
  const float* w = W + threadIdx.x * DIM;
  float s = 0.f;
  for (int k = 0; k < DIM; ++k) s = fmaf(l[k], w[k], s);
  out[bb * DIM + threadIdx.x] = s;
}

// ---------------- kernel 3: top-16 kNN, one wave32 per query ----------------
__global__ void knn_kernel(const float* __restrict__ xyz_q, const float* __restrict__ xyz,
                           int* __restrict__ knn_out) {
  __shared__ float s_xyz[NPTS * 3];
  __shared__ float s_cd[8][KNN * 32];
  __shared__ int   s_ci[8][KNN * 32];
  const int tid  = threadIdx.x;
  const int bidx = blockIdx.x / (NQ / 8);
  const int q0   = (blockIdx.x % (NQ / 8)) * 8;
  // async stage the xyz tile (24 KB, contiguous) into LDS
  {
    const float* src = xyz + (size_t)bidx * NPTS * 3;
    for (int c = tid; c < NPTS * 3 / 4; c += 256)
      async_copy_b128(&s_xyz[c * 4], src + c * 4);
    wait_async0();
  }
  __syncthreads();
  const int wave = tid >> 5, lane = tid & 31;
  const int qg = q0 + wave;
  const float* qp = xyz_q + ((size_t)bidx * NQ + qg) * 3;
  const float qx = qp[0], qy = qp[1], qz = qp[2];
  float dist[KNN]; int idxs[KNN];
#pragma unroll
  for (int t = 0; t < KNN; ++t) { dist[t] = 3.4e38f; idxs[t] = 0; }
  for (int p = lane; p < NPTS; p += 32) {
    float dx = s_xyz[p * 3 + 0] - qx, dy = s_xyz[p * 3 + 1] - qy, dz = s_xyz[p * 3 + 2] - qz;
    float d2 = fmaf(dx, dx, fmaf(dy, dy, dz * dz));
    if (d2 < dist[KNN - 1]) {        // sorted-insert swap chain (ascending)
      float v = d2; int vi = p;
#pragma unroll
      for (int t = 0; t < KNN; ++t) {
        if (v < dist[t]) { float td = dist[t]; dist[t] = v; v = td;
                           int   ti = idxs[t]; idxs[t] = vi; vi = ti; }
      }
    }
  }
#pragma unroll
  for (int t = 0; t < KNN; ++t) { s_cd[wave][lane * KNN + t] = dist[t]; s_ci[wave][lane * KNN + t] = idxs[t]; }
  __syncthreads();
  // 32-way merge of per-lane sorted lists: 16 rounds of wave argmin
  int ptr = 0;
  for (int r = 0; r < KNN; ++r) {
    float v = (ptr < KNN) ? s_cd[wave][lane * KNN + ptr] : 3.4e38f;
    int  pi = (ptr < KNN) ? s_ci[wave][lane * KNN + ptr] : -1;
    int  ln = lane;
#pragma unroll
    for (int off = 16; off >= 1; off >>= 1) {
      float ov = __shfl_xor(v, off);
      int  oln = __shfl_xor(ln, off);
      int  opi = __shfl_xor(pi, off);
      if (ov < v || (ov == v && oln < ln)) { v = ov; ln = oln; pi = opi; }
    }
    if (lane == 0) knn_out[((size_t)bidx * NQ + qg) * KNN + r] = pi;
    if (lane == ln) ptr++;
  }
}

// ---------------- kernel 4: K = points@Wk^T, V = points@Wv^T (WMMA) ----------------
__global__ void proj_kernel(const bf16* __restrict__ pointsb, const bf16* __restrict__ Wkb,
                            const bf16* __restrict__ Wvb, float* __restrict__ Kp, float* __restrict__ Vp) {
  __shared__ bf16 sA[64 * DIM];
  const int tid  = threadIdx.x;
  const int bidx = blockIdx.x / (NPTS / 64);
  const int m0   = (blockIdx.x % (NPTS / 64)) * 64;
  // async stage the 32 KB bf16 A-tile (row-major == WMMA A staging layout)
  {
    const bf16* src = pointsb + ((size_t)bidx * NPTS + m0) * DIM;
    for (int c = tid; c < 64 * DIM / 8; c += 256)
      async_copy_b128(&sA[c * 8], src + c * 8);
    wait_async0();
  }
  __syncthreads();
  const int wave = tid >> 5, lane = tid & 31;
  // wave owns Nt = wave, wave+8; B fragments preloaded once per (Nt, weight)
  // and reused across all 4 M-tiles; M loop kept rolled to bound VGPR pressure.
#pragma unroll 1
  for (int half = 0; half < 2; ++half) {
    const int Nt = wave + 8 * half;
    const int n  = Nt * 16 + (lane & 15);
#pragma unroll 1
    for (int mat = 0; mat < 2; ++mat) {
      const bf16* W = mat ? Wvb : Wkb;
      float* O = mat ? Vp : Kp;
      v16bf bfr[8];
#pragma unroll
      for (int k8 = 0; k8 < 8; ++k8) bfr[k8] = load_frag(W, Nt * 16, k8 * 32);
#pragma unroll 1
      for (int mp = 0; mp < 2; ++mp) {            // M-tile pairs {0,1},{2,3}
        v8f a0 = {}, a1 = {};
#pragma unroll
        for (int k8 = 0; k8 < 8; ++k8) {
          a0 = wmma_acc(load_frag(sA, mp * 32,      k8 * 32), bfr[k8], a0);
          a1 = wmma_acc(load_frag(sA, mp * 32 + 16, k8 * 32), bfr[k8], a1);
        }
        int mb = m0 + mp * 32 + 8 * (lane >> 4);
#pragma unroll
        for (int r = 0; r < 8; ++r) {
          O[((size_t)bidx * NPTS + mb + r) * DIM + n]      = a0[r];
          O[((size_t)bidx * NPTS + mb + 16 + r) * DIM + n] = a1[r];
        }
      }
    }
  }
}

// ---------------- kernel 5: fused pos-MLP + gamma-MLP + softmax + output ----------------
// block = 256 threads (8 waves), QT=4 queries, one thread per feature column.
#define SMEM_IDX   0
#define SMEM_D     256
#define SMEM_P     1024                       // bf16 [QT*17][256]  = 34816 B
#define SMEM_X     35840                      // bf16 [80][256]     = 40960 B
#define SMEM_G     76800                      // bf16 [80][256]     = 40960 B
#define SMEM_LOG   117760                     // f32  [68][256]     = 69632 B
#define SMEM_TOTAL 187392

__global__ void fused_attn_kernel(const float* __restrict__ xyz_q, const float* __restrict__ xyz,
                                  const float* __restrict__ Wd1, const float* __restrict__ bd1,
                                  const float* __restrict__ bd2, const float* __restrict__ bg1,
                                  const float* __restrict__ bg2, const bf16* __restrict__ Wd2b,
                                  const bf16* __restrict__ Wg1b, const bf16* __restrict__ Wg2b,
                                  const int* __restrict__ knn_idx, const float* __restrict__ Kp,
                                  const float* __restrict__ Vp, const float* __restrict__ qa,
                                  const float* __restrict__ kgv, const float* __restrict__ vgv,
                                  float* __restrict__ out) {
  extern __shared__ char smem[];
  int*   sIdx = (int*)(smem + SMEM_IDX);
  float* sD   = (float*)(smem + SMEM_D);
  bf16*  sP   = (bf16*)(smem + SMEM_P);
  bf16*  sX   = (bf16*)(smem + SMEM_X);
  bf16*  sG   = (bf16*)(smem + SMEM_G);
  float* sLog = (float*)(smem + SMEM_LOG);

  const int tid  = threadIdx.x;
  const int wave = tid >> 5, lane = tid & 31;
  const int bidx = blockIdx.x / (NQ / QT);
  const int q0   = (blockIdx.x % (NQ / QT)) * QT;

  if (tid < QT * KNN) sIdx[tid] = knn_idx[((size_t)bidx * NQ + q0) * KNN + tid];
  __syncthreads();

  // relative offsets d = xyz_q - xyz[idx]; zero the global-token pos slot (j==16)
  for (int t = tid; t < QT * KNN * 3; t += 256) {
    int q = t / (KNN * 3), rem = t % (KNN * 3), j = rem / 3, c = rem % 3;
    sD[t] = xyz_q[((size_t)bidx * NQ + q0 + q) * 3 + c]
          - xyz[((size_t)bidx * NPTS + sIdx[q * KNN + j]) * 3 + c];
  }
  for (int q = 0; q < QT; ++q) sP[(q * 17 + 16) * DIM + tid] = tobf(0.f);
  __syncthreads();

  // stage 1 (VALU, K=3): H1 = relu(d @ Wd1^T + bd1) -> sX rows 0..63
  {
    float w0 = Wd1[tid * 3 + 0], w1 = Wd1[tid * 3 + 1], w2 = Wd1[tid * 3 + 2], bb = bd1[tid];
    for (int r = 0; r < QT * KNN; ++r) {
      float v = fmaf(sD[r * 3 + 0], w0, fmaf(sD[r * 3 + 1], w1, fmaf(sD[r * 3 + 2], w2, bb)));
      sX[r * DIM + tid] = tobf(fmaxf(v, 0.f));
    }
  }
  __syncthreads();

  // stage 2 (WMMA): pos = H1 @ Wd2^T + bd2 -> sP[(q*17+j)]
  // wave owns Nt = wave, wave+8; 8 B-fragments preloaded once per Nt,
  // M-tile pair loop kept rolled (2 accumulators) -> no spills.
#pragma unroll 1
  for (int half = 0; half < 2; ++half) {
    const int Nt = wave + 8 * half;
    const int n  = Nt * 16 + (lane & 15);
    const float bb = bd2[n];
    v16bf bfr[8];
#pragma unroll
    for (int k8 = 0; k8 < 8; ++k8) bfr[k8] = load_frag(Wd2b, Nt * 16, k8 * 32);
#pragma unroll 1
    for (int mp = 0; mp < 2; ++mp) {              // M-tiles {0,1},{2,3}
      v8f a0 = {}, a1 = {};
#pragma unroll
      for (int k8 = 0; k8 < 8; ++k8) {
        a0 = wmma_acc(load_frag(sX, mp * 32,      k8 * 32), bfr[k8], a0);
        a1 = wmma_acc(load_frag(sX, mp * 32 + 16, k8 * 32), bfr[k8], a1);
      }
      int mb = mp * 32 + 8 * (lane >> 4);
#pragma unroll
      for (int r = 0; r < 8; ++r) {
        int m = mb + r, q = m >> 4, j = m & 15;
        sP[(q * 17 + j) * DIM + n] = tobf(a0[r] + bb);
        int m2 = m + 16; q = m2 >> 4; j = m2 & 15;
        sP[(q * 17 + j) * DIM + n] = tobf(a1[r] + bb);
      }
    }
  }
  __syncthreads();

  // stage 3: X = q_attn - k + pos (rows q*17+j, bf16), pad to 80 rows
  {
    float qav = qa[bidx * DIM + tid];
    float kgl = kgv[bidx * DIM + tid];
    for (int q = 0; q < QT; ++q)
      for (int j = 0; j < 17; ++j) {
        int r = q * 17 + j;
        float kv = (j < KNN) ? Kp[((size_t)bidx * NPTS + sIdx[q * KNN + j]) * DIM + tid] : kgl;
        sX[r * DIM + tid] = tobf(qav - kv + bf2f(sP[r * DIM + tid]));
      }
    for (int r = QT * 17; r < 80; ++r) sX[r * DIM + tid] = tobf(0.f);
  }
  __syncthreads();

  // stage 4 (WMMA): G = relu(X @ Wg1^T + bg1); 5 M-tiles = 2 pairs + 1 single
#pragma unroll 1
  for (int half = 0; half < 2; ++half) {
    const int Nt = wave + 8 * half;
    const int n  = Nt * 16 + (lane & 15);
    const float bb = bg1[n];
    v16bf bfr[8];
#pragma unroll
    for (int k8 = 0; k8 < 8; ++k8) bfr[k8] = load_frag(Wg1b, Nt * 16, k8 * 32);
#pragma unroll 1
    for (int mp = 0; mp < 2; ++mp) {
      v8f a0 = {}, a1 = {};
#pragma unroll
      for (int k8 = 0; k8 < 8; ++k8) {
        a0 = wmma_acc(load_frag(sX, mp * 32,      k8 * 32), bfr[k8], a0);
        a1 = wmma_acc(load_frag(sX, mp * 32 + 16, k8 * 32), bfr[k8], a1);
      }
      int mb = mp * 32 + 8 * (lane >> 4);
#pragma unroll
      for (int r = 0; r < 8; ++r) {
        sG[(mb + r) * DIM + n]      = tobf(fmaxf(a0[r] + bb, 0.f));
        sG[(mb + 16 + r) * DIM + n] = tobf(fmaxf(a1[r] + bb, 0.f));
      }
    }
    {                                              // tail M-tile (rows 64..79)
      v8f a0 = {};
#pragma unroll
      for (int k8 = 0; k8 < 8; ++k8)
        a0 = wmma_acc(load_frag(sX, 64, k8 * 32), bfr[k8], a0);
      int mb = 64 + 8 * (lane >> 4);
#pragma unroll
      for (int r = 0; r < 8; ++r)
        sG[(mb + r) * DIM + n] = tobf(fmaxf(a0[r] + bb, 0.f));
    }
  }
  __syncthreads();

  // stage 5 (WMMA): logits = G @ Wg2^T + bg2 (f32 in LDS for softmax)
#pragma unroll 1
  for (int half = 0; half < 2; ++half) {
    const int Nt = wave + 8 * half;
    const int n  = Nt * 16 + (lane & 15);
    const float bb = bg2[n];
    v16bf bfr[8];
#pragma unroll
    for (int k8 = 0; k8 < 8; ++k8) bfr[k8] = load_frag(Wg2b, Nt * 16, k8 * 32);
#pragma unroll 1
    for (int mp = 0; mp < 2; ++mp) {
      v8f a0 = {}, a1 = {};
#pragma unroll
      for (int k8 = 0; k8 < 8; ++k8) {
        a0 = wmma_acc(load_frag(sG, mp * 32,      k8 * 32), bfr[k8], a0);
        a1 = wmma_acc(load_frag(sG, mp * 32 + 16, k8 * 32), bfr[k8], a1);
      }
      int mb = mp * 32 + 8 * (lane >> 4);
#pragma unroll
      for (int r = 0; r < 8; ++r) {
        sLog[(mb + r) * DIM + n]      = a0[r] + bb;
        sLog[(mb + 16 + r) * DIM + n] = a1[r] + bb;
      }
    }
    {                                              // tail M-tile: rows 64..67 only
      v8f a0 = {};
#pragma unroll
      for (int k8 = 0; k8 < 8; ++k8)
        a0 = wmma_acc(load_frag(sG, 64, k8 * 32), bfr[k8], a0);
      int mb = 64 + 8 * (lane >> 4);
#pragma unroll
      for (int r = 0; r < 8; ++r) {
        int m = mb + r;
        if (m < QT * 17) sLog[m * DIM + n] = a0[r] + bb;
      }
    }
  }
  __syncthreads();

  // stage 6: softmax over the 17 neighbors + weighted sum of (v + pos)
  {
    float vgl = vgv[bidx * DIM + tid];
    for (int q = 0; q < QT; ++q) {
      float mx = -3.4e38f;
      for (int j = 0; j < 17; ++j) mx = fmaxf(mx, sLog[(q * 17 + j) * DIM + tid]);
      float s = 0.f, accv = 0.f;
      for (int j = 0; j < 17; ++j) {
        float e  = __expf(sLog[(q * 17 + j) * DIM + tid] - mx);
        float vv = (j < KNN) ? Vp[((size_t)bidx * NPTS + sIdx[q * KNN + j]) * DIM + tid] : vgl;
        accv = fmaf(e, vv + bf2f(sP[(q * 17 + j) * DIM + tid]), accv);
        s += e;
      }
      out[((size_t)bidx * NQ + q0 + q) * DIM + tid] = accv / s;
    }
  }
}

// ---------------- host ----------------
extern "C" void kernel_launch(void* const* d_in, const int* in_sizes, int n_in,
                              void* d_out, int out_size, void* d_ws, size_t ws_size,
                              hipStream_t stream) {
  const float* xyz_q = (const float*)d_in[0];
  const float* lat   = (const float*)d_in[1];
  const float* xyz   = (const float*)d_in[2];
  const float* pts   = (const float*)d_in[3];
  const float* Wd1 = (const float*)d_in[4];  const float* bd1 = (const float*)d_in[5];
  const float* Wd2 = (const float*)d_in[6];  const float* bd2 = (const float*)d_in[7];
  const float* Wg1 = (const float*)d_in[8];  const float* bg1 = (const float*)d_in[9];
  const float* Wg2 = (const float*)d_in[10]; const float* bg2 = (const float*)d_in[11];
  const float* Wkg = (const float*)d_in[12];
  const float* Wvg = (const float*)d_in[13];
  const float* Wq  = (const float*)d_in[14];
  const float* Wk  = (const float*)d_in[15];
  const float* Wv  = (const float*)d_in[16];
  float* out = (float*)d_out;

  // workspace carve-up (~22.7 MB total)
  char* ws = (char*)d_ws;
  size_t off = 0;
  int*   d_knn = (int*)(ws + off);  off += (size_t)BATCH * NQ * KNN * sizeof(int);
  float* d_Kp  = (float*)(ws + off); off += (size_t)BATCH * NPTS * DIM * sizeof(float);
  float* d_Vp  = (float*)(ws + off); off += (size_t)BATCH * NPTS * DIM * sizeof(float);
  bf16*  d_wb  = (bf16*)(ws + off);  off += (size_t)5 * DIM * DIM * sizeof(bf16);
  float* d_qa  = (float*)(ws + off); off += (size_t)BATCH * DIM * sizeof(float);
  float* d_kg  = (float*)(ws + off); off += (size_t)BATCH * DIM * sizeof(float);
  float* d_vg  = (float*)(ws + off); off += (size_t)BATCH * DIM * sizeof(float);
  bf16*  d_pb  = (bf16*)(ws + off);  off += (size_t)BATCH * NPTS * DIM * sizeof(bf16);
  bf16* Wd2b = d_wb + 0 * DIM * DIM;
  bf16* Wg1b = d_wb + 1 * DIM * DIM;
  bf16* Wg2b = d_wb + 2 * DIM * DIM;
  bf16* Wkb  = d_wb + 3 * DIM * DIM;
  bf16* Wvb  = d_wb + 4 * DIM * DIM;

  conv_bf16_kernel<<<2048, 256, 0, stream>>>(Wd2, Wg1, Wg2, Wk, Wv, pts, d_wb, d_pb);
  small_proj_kernel<<<12, 256, 0, stream>>>(lat, Wq, Wkg, Wvg, d_qa, d_kg, d_vg);
  knn_kernel<<<BATCH * (NQ / 8), 256, 0, stream>>>(xyz_q, xyz, d_knn);
  proj_kernel<<<BATCH * (NPTS / 64), 256, 0, stream>>>(d_pb, Wkb, Wvb, d_Kp, d_Vp);
  fused_attn_kernel<<<BATCH * (NQ / QT), 256, SMEM_TOTAL, stream>>>(
      xyz_q, xyz, Wd1, bd1, bd2, bg1, bg2, Wd2b, Wg1b, Wg2b,
      d_knn, d_Kp, d_Vp, d_qa, d_kg, d_vg, out);
}
